// RDLNetSAMEncoder_8409545965964
// MI455X (gfx1250) — matrix operand order repaired and checked by reading, not compiled, not run on hardware
//
#include <hip/hip_runtime.h>
#include <stdint.h>

typedef __attribute__((ext_vector_type(16))) _Float16 v16h;
typedef __attribute__((ext_vector_type(8)))  _Float16 v8h;
typedef __attribute__((ext_vector_type(8)))  float    v8f;
typedef __attribute__((ext_vector_type(4)))  int      v4i;

#define BQ    2
#define GRD   32
#define TOK   1024          /* tokens per batch  */
#define ROWS  2048          /* BQ*TOK            */
#define DIMC  384
#define HEADS 8
#define HD    48
#define HDP   64            /* padded head dim for WMMA K */
#define MLPC  1536
#define DEPTH 12
#define WSZ   14
#define NWIN  9             /* windows per batch (42/14)^2 */
#define WTOK  196           /* tokens per window */
#define WTOKP 224           /* WTOK padded to a multiple of 32 */

// ---------------------------------------------------------------------------
// gfx1250 async global->LDS copies (ASYNCcnt path). Guarded so compile never
// breaks if the toolchain lacks the builtins; fallback is vector ld/st.
// Builtin signature (from hipcc diagnostic): param0 = AS1 v4i*, so pass
// address-space-qualified int4 pointers.
// ---------------------------------------------------------------------------
#if defined(__gfx1250__) && __has_builtin(__builtin_amdgcn_global_load_async_to_lds_b128)
#define USE_ASYNC_LDS 1
#else
#define USE_ASYNC_LDS 0
#endif

typedef __attribute__((address_space(1))) v4i* gptr_v4i;   // global int4*
typedef __attribute__((address_space(3))) v4i* lptr_v4i;   // LDS int4*

__device__ __forceinline__ void async_wait_all() {
#if USE_ASYNC_LDS
#if __has_builtin(__builtin_amdgcn_s_wait_asynccnt)
  __builtin_amdgcn_s_wait_asynccnt(0);
#else
  asm volatile("s_wait_asynccnt 0" ::: "memory");
#endif
#endif
}

__device__ __forceinline__ void cp16_g2l(const _Float16* src, _Float16* dstLds) {
#if USE_ASYNC_LDS
  __builtin_amdgcn_global_load_async_to_lds_b128(
      (gptr_v4i)(const_cast<_Float16*>(src)), (lptr_v4i)(dstLds), 0, 0);
#else
  *(v8h*)dstLds = *(const v8h*)src;
#endif
}

// ---------------------------------------------------------------------------
// WMMA fragment loaders (16-bit A 16x32 / B 32x16 lane layouts per CDNA5 ISA)
// ---------------------------------------------------------------------------
__device__ __forceinline__ v16h frag_a(const _Float16* p) {
  // lane<16: K=0..7 then 16..23 ; lane>=16: K=8..15 then 24..31 (caller offsets p)
  v8h lo = *(const v8h*)p;
  v8h hi = *(const v8h*)(p + 16);
  v16h r;
#pragma unroll
  for (int i = 0; i < 8; ++i) { r[i] = lo[i]; r[8 + i] = hi[i]; }
  return r;
}
__device__ __forceinline__ v16h frag_b(const _Float16* p) {
  // lane<16: K=0..7 then 16..23 ; lane>=16: K=8..15 then 24..31 (caller offsets p)
  v8h lo = *(const v8h*)p;
  v8h hi = *(const v8h*)(p + 16);
  v16h r;
#pragma unroll
  for (int i = 0; i < 8; ++i) { r[i] = lo[i]; r[8 + i] = hi[i]; }
  return r;
}

// ---------------------------------------------------------------------------
// Batched GEMM: C[z] = act(alpha * A[z](MxK,f16) * Bt[z](NxK,f16)^T + bias) + res
// Requirements (arranged by host): K % 32 == 0, lda/ldb % 8 == 0.
// Tile 128x64x32, 256 threads (8 wave32), wave tile 32x32 = 2x2 WMMA.
// Staging is pure 128-bit traffic: async global->LDS on interior tiles,
// clamped vector loads + zero-select on M/N edge tiles (no exec-mask churn).
// ---------------------------------------------------------------------------
__global__ __launch_bounds__(256) void gemm_wmma(
    const _Float16* __restrict__ A,  long long sA,  int lda,
    const _Float16* __restrict__ Bt, long long sB,  int ldb,
    float* __restrict__ C,           long long sC,  int ldc,
    _Float16* __restrict__ C16,      long long sC16, int ldc16,
    const float* __restrict__ bias,
    const float* __restrict__ res,
    int M, int N, int K, float alpha, int act)
{
  __shared__ __align__(16) _Float16 As[128][40];
  __shared__ __align__(16) _Float16 Bs[64][40];

  const int z = blockIdx.z;
  A  += (long long)z * sA;
  Bt += (long long)z * sB;
  C  += (long long)z * sC;
  if (C16) C16 += (long long)z * sC16;

  const int bm   = blockIdx.y * 128;
  const int bn   = blockIdx.x * 64;
  const int tid  = threadIdx.x;
  const int wid  = tid >> 5;
  const int lane = tid & 31;
  const int wm   = (wid >> 1) * 32;
  const int wn   = (wid & 1) * 32;
  const int lr   = lane & 15;
  const bool hi  = lane >= 16;

  const bool intA = (bm + 128 <= M);   // whole A tile in-bounds
  const bool intB = (bn + 64 <= N);    // whole B tile in-bounds

  // per-thread chunk coordinates (16B chunks of a 32-half K-slice)
  const int ra0 = tid >> 2,           ca0 = (tid & 3) * 8;        // A chunk 0
  const int ra1 = (tid + 256) >> 2,   ca1 = ca0;                  // A chunk 1
  const int rb  = tid >> 2,           cb  = (tid & 3) * 8;        // B chunk

  v8f acc00 = {}, acc01 = {}, acc10 = {}, acc11 = {};

  for (int k0 = 0; k0 < K; k0 += 32) {
    if (intA) {
      cp16_g2l(A + (long long)(bm + ra0) * lda + k0 + ca0, &As[ra0][ca0]);
      cp16_g2l(A + (long long)(bm + ra1) * lda + k0 + ca1, &As[ra1][ca1]);
    } else {
#pragma unroll
      for (int i = 0; i < 2; ++i) {
        int r = i ? ra1 : ra0, c8 = i ? ca1 : ca0;
        int gm = bm + r;
        v8h v = *(const v8h*)(A + (long long)(gm < M ? gm : 0) * lda + k0 + c8);
        if (gm >= M) { v8h zz = {}; v = zz; }
        *(v8h*)&As[r][c8] = v;
      }
    }
    if (intB) {
      cp16_g2l(Bt + (long long)(bn + rb) * ldb + k0 + cb, &Bs[rb][cb]);
    } else {
      int gn = bn + rb;
      v8h v = *(const v8h*)(Bt + (long long)(gn < N ? gn : 0) * ldb + k0 + cb);
      if (gn >= N) { v8h zz = {}; v = zz; }
      *(v8h*)&Bs[rb][cb] = v;
    }
    async_wait_all();
    __syncthreads();

    const int kh = hi ? 8 : 0;   // A frag: half offset within 32-K tile
    const int kb = hi ? 8 : 0;   // B frag: half offset within 32-K tile
    v16h a0 = frag_a(&As[wm + lr][kh]);
    v16h a1 = frag_a(&As[wm + 16 + lr][kh]);
    v16h b0 = frag_b(&Bs[wn + lr][kb]);
    v16h b1 = frag_b(&Bs[wn + 16 + lr][kb]);
    acc00 = __builtin_amdgcn_wmma_f32_16x16x32_f16(false, a0, false, b0, (short)0, acc00, false, false);
    acc01 = __builtin_amdgcn_wmma_f32_16x16x32_f16(false, a0, false, b1, (short)0, acc01, false, false);
    acc10 = __builtin_amdgcn_wmma_f32_16x16x32_f16(false, a1, false, b0, (short)0, acc10, false, false);
    acc11 = __builtin_amdgcn_wmma_f32_16x16x32_f16(false, a1, false, b1, (short)0, acc11, false, false);
    __syncthreads();
  }

  v8f* accs[2][2] = {{&acc00, &acc01}, {&acc10, &acc11}};
#pragma unroll
  for (int i = 0; i < 2; ++i)
#pragma unroll
    for (int j = 0; j < 2; ++j) {
      int nn = bn + wn + j * 16 + lr;
      if (nn >= N) continue;
      float bv = bias ? bias[nn] : 0.f;
      v8f a = *accs[i][j];
#pragma unroll
      for (int r = 0; r < 8; ++r) {
        int mm = bm + wm + i * 16 + (hi ? 8 : 0) + r;
        if (mm >= M) continue;
        float v = alpha * a[r] + bv;
        if (act) v = 0.5f * v * (1.f + erff(v * 0.70710678118654752f));   // exact GELU
        if (res) v += res[(long long)mm * ldc + nn];
        C[(long long)mm * ldc + nn] = v;
        if (C16) C16[(long long)mm * ldc16 + nn] = (_Float16)v;
      }
    }
}

// ---------------------------------------------------------------------------
// LayerNorm (fp32 in -> f16 out for GEMM consumption). One row per block.
// ---------------------------------------------------------------------------
__global__ void ln_kernel(const float* __restrict__ x, const float* __restrict__ g,
                          const float* __restrict__ b, _Float16* __restrict__ out, int C)
{
  long long row = blockIdx.x;
  const float* xr = x + row * C;
  __shared__ float red[128];
  int tid = threadIdx.x;
  float s = 0.f;
  for (int c = tid; c < C; c += 128) s += xr[c];
  red[tid] = s; __syncthreads();
  for (int off = 64; off > 0; off >>= 1) { if (tid < off) red[tid] += red[tid + off]; __syncthreads(); }
  float mu = red[0] / C;
  __syncthreads();
  float v = 0.f;
  for (int c = tid; c < C; c += 128) { float d = xr[c] - mu; v += d * d; }
  red[tid] = v; __syncthreads();
  for (int off = 64; off > 0; off >>= 1) { if (tid < off) red[tid] += red[tid + off]; __syncthreads(); }
  float rstd = rsqrtf(red[0] / C + 1e-6f);
  for (int c = tid; c < C; c += 128)
    out[row * C + c] = (_Float16)((xr[c] - mu) * rstd * g[c] + b[c]);
}

// ---------------------------------------------------------------------------
// Weight fp32 [K][N] -> f16 [N][K] (transposed for GEMM Bt)
// ---------------------------------------------------------------------------
__global__ void wconv_t(const float* __restrict__ w, _Float16* __restrict__ out,
                        int K, int N)
{
  long long i = (long long)blockIdx.x * 256 + threadIdx.x;
  if (i >= (long long)N * K) return;
  int n = (int)(i / K), k = (int)(i % K);
  out[i] = (_Float16)w[(long long)k * N + n];
}
__global__ void wconv_copy(const float* __restrict__ w, _Float16* __restrict__ out, long long n)
{
  long long i = (long long)blockIdx.x * 256 + threadIdx.x;
  if (i < n) out[i] = (_Float16)w[i];
}
__global__ void fill_h16(_Float16* __restrict__ p, long long n)
{
  long long i = (long long)blockIdx.x * 256 + threadIdx.x;
  if (i < n) p[i] = (_Float16)0.f;
}

// im2col for 16x16/16 patch conv:  A16[b*1024 + py*32+px][c*256+iy*16+ix]
__global__ void im2col_patch(const float* __restrict__ x, _Float16* __restrict__ a16)
{
  int row = blockIdx.x;
  int b = row >> 10, t = row & 1023, py = t >> 5, px = t & 31;
  for (int col = threadIdx.x; col < 768; col += 256) {
    int c = col >> 8, rem = col & 255, iy = rem >> 4, ix = rem & 15;
    a16[(long long)row * 768 + col] =
        (_Float16)x[(((long long)b * 3 + c) * 512 + py * 16 + iy) * 512 + px * 16 + ix];
  }
}

__global__ void add_pos(float* __restrict__ x, const float* __restrict__ pos)
{
  long long i = (long long)blockIdx.x * 256 + threadIdx.x;
  if (i < (long long)ROWS * DIMC) x[i] += pos[i % ((long long)TOK * DIMC)];
}

// ---------------------------------------------------------------------------
// qkv split, global attention.  q16/k16: [bh=16][1024][64] (d>=48 zero pad).
// v16t: [bh][64][Tp] (row/col pads pre-zeroed by fill_h16).
// ---------------------------------------------------------------------------
__global__ void split_qkv_global(const float* __restrict__ qkv,
                                 _Float16* __restrict__ q16, _Float16* __restrict__ k16,
                                 _Float16* __restrict__ v16t, int Tp)
{
  long long i = (long long)blockIdx.x * 256 + threadIdx.x;
  if (i >= 16LL * TOK * HDP) return;
  int d = (int)(i & 63);
  long long r = i >> 6;
  int t = (int)(r & 1023), bh = (int)(r >> 10);
  int b = bh >> 3, h = bh & 7;
  const float* src = qkv + (long long)(b * TOK + t) * (3 * DIMC) + h * HD;
  q16[i] = (d < HD) ? (_Float16)src[d]        : (_Float16)0.f;
  k16[i] = (d < HD) ? (_Float16)src[DIMC + d] : (_Float16)0.f;
  if (d < HD) v16t[((long long)bh * HDP + d) * Tp + t] = (_Float16)src[2 * DIMC + d];
}

// qkv split: windowed (pad 32->42, 3x3 windows of 14). Pad tokens attend with
// bias-only qkv (LN output is zero there), matching the reference's zero-pad.
__global__ void split_qkv_win(const float* __restrict__ qkv, const float* __restrict__ qkv_b,
                              _Float16* __restrict__ q16, _Float16* __restrict__ k16,
                              _Float16* __restrict__ v16t, int Tp)
{
  long long i = (long long)blockIdx.x * 256 + threadIdx.x;
  if (i >= 144LL * WTOK * HDP) return;
  int d = (int)(i & 63);
  long long r = i >> 6;
  int t = (int)(r % WTOK), bh = (int)(r / WTOK);
  int h = bh & 7, w = bh >> 3;
  int b = w / NWIN, wi = w % NWIN, wy = wi / 3, wx = wi % 3;
  int rr = t / WSZ, cc = t % WSZ;
  int gy = wy * WSZ + rr, gx = wx * WSZ + cc;
  int off = h * HD;
  float qf = 0.f, kf = 0.f, vf = 0.f;
  if (d < HD) {
    if (gy < GRD && gx < GRD) {
      const float* src = qkv + (long long)(b * TOK + gy * GRD + gx) * (3 * DIMC) + off;
      qf = src[d]; kf = src[DIMC + d]; vf = src[2 * DIMC + d];
    } else {
      qf = qkv_b[off + d]; kf = qkv_b[DIMC + off + d]; vf = qkv_b[2 * DIMC + off + d];
    }
  }
  q16[i] = (_Float16)qf;
  k16[i] = (_Float16)kf;
  if (d < HD) v16t[((long long)bh * HDP + d) * Tp + t] = (_Float16)vf;
}

// rel-pos bias: out[bh*T + t][k] = sum_c q[bh][t][c] * relTab[coord - k + S - 1][c]
__global__ void rel_bias(const _Float16* __restrict__ q16, const float* __restrict__ relTab,
                         float* __restrict__ out, int T, int W, int S, int coordIsY)
{
  long long blk = blockIdx.x;
  int t = (int)(blk % T);
  long long bh = blk / T;
  int y = t / W, x = t % W;
  int coord = coordIsY ? y : x;
  __shared__ float qrow[HD];
  const _Float16* qp = q16 + (bh * T + t) * HDP;
  for (int c = threadIdx.x; c < HD; c += 32) qrow[c] = (float)qp[c];
  __syncthreads();
  for (int k = threadIdx.x; k < S; k += 32) {
    const float* R = relTab + (long long)(coord - k + S - 1) * HD;
    float s = 0.f;
#pragma unroll 8
    for (int c = 0; c < HD; ++c) s += qrow[c] * R[c];
    out[blk * S + k] = s;
  }
}

// fused rel-pos bias add + softmax; writes f16 probs with padded row stride Tp
__global__ void softmax_bias(float* __restrict__ attn, _Float16* __restrict__ attn16,
                             const float* __restrict__ relh, const float* __restrict__ relw,
                             int T, int Tp, int W, int S)
{
  long long row = blockIdx.x;
  int tid = threadIdx.x;
  float* ap = attn + row * (long long)T;
  _Float16* op = attn16 + row * (long long)Tp;
  const float* rh = relh + row * (long long)S;
  const float* rw = relw + row * (long long)S;
  __shared__ float red[256];
  float mx = -1e30f;
  for (int j = tid; j < T; j += 256) {
    float t = ap[j] + rh[j / W] + rw[j % W];
    ap[j] = t;
    mx = fmaxf(mx, t);
  }
  red[tid] = mx; __syncthreads();
  for (int off = 128; off > 0; off >>= 1) { if (tid < off) red[tid] = fmaxf(red[tid], red[tid + off]); __syncthreads(); }
  mx = red[0]; __syncthreads();
  float s = 0.f;
  for (int j = tid; j < T; j += 256) s += expf(ap[j] - mx);
  red[tid] = s; __syncthreads();
  for (int off = 128; off > 0; off >>= 1) { if (tid < off) red[tid] += red[tid + off]; __syncthreads(); }
  float inv = 1.f / red[0];
  for (int j = tid; j < Tp; j += 256)
    op[j] = (j < T) ? (_Float16)(expf(ap[j] - mx) * inv) : (_Float16)0.f;
}

// merge heads back to [b][1024][384] f16 (proj GEMM input); ctx row stride 64
__global__ void merge_global(const float* __restrict__ ctx, _Float16* __restrict__ merged)
{
  long long i = (long long)blockIdx.x * 256 + threadIdx.x;
  if (i >= (long long)ROWS * DIMC) return;
  int c = (int)(i % DIMC);
  long long r = i / DIMC;
  int t = (int)(r & 1023), b = (int)(r >> 10);
  int h = c / HD, d = c % HD;
  merged[i] = (_Float16)ctx[(((long long)(b * HEADS + h)) * TOK + t) * HDP + d];
}
__global__ void merge_win(const float* __restrict__ ctx, _Float16* __restrict__ merged)
{
  long long i = (long long)blockIdx.x * 256 + threadIdx.x;
  if (i >= (long long)ROWS * DIMC) return;
  int c = (int)(i % DIMC);
  long long r = i / DIMC;
  int t = (int)(r & 1023), b = (int)(r >> 10);
  int gy = t >> 5, gx = t & 31;
  int wy = gy / WSZ, rr = gy % WSZ, wx = gx / WSZ, cc = gx % WSZ;
  int w = b * NWIN + wy * 3 + wx;
  int h = c / HD, d = c % HD;
  int tw = rr * WSZ + cc;
  merged[i] = (_Float16)ctx[(((long long)(w * HEADS + h)) * WTOK + tw) * HDP + d];
}

__global__ void copyf(const float* __restrict__ src, float* __restrict__ dst, long long n)
{
  long long i = (long long)blockIdx.x * 256 + threadIdx.x;
  if (i < n) dst[i] = src[i];
}

// ---------------------------------------------------------------------------
extern "C" void kernel_launch(void* const* d_in, const int* in_sizes, int n_in,
                              void* d_out, int out_size, void* d_ws, size_t ws_size,
                              hipStream_t stream)
{
  (void)in_sizes; (void)n_in; (void)out_size; (void)ws_size;

  // Inputs in setup_inputs() insertion order:
  // x, patch_w, patch_b, pos_embed, then per block:
  // ln1_g, ln1_b, qkv_w, qkv_b, proj_w, proj_b, rel_h, rel_w, ln2_g, ln2_b,
  // fc1_w, fc1_b, fc2_w, fc2_b
  const float* X       = (const float*)d_in[0];
  const float* patch_w = (const float*)d_in[1];
  const float* patch_b = (const float*)d_in[2];
  const float* pos     = (const float*)d_in[3];
  auto blkP = [&](int i, int j) { return (const float*)d_in[4 + i * 14 + j]; };

  float* out = (float*)d_out;
  const long long OUTN = (long long)ROWS * DIMC;   // 786432 per tensor

  // -------- workspace bump allocator --------
  uintptr_t wsp = (uintptr_t)d_ws;
  auto alloc = [&](size_t bytes) -> void* {
    uintptr_t p = (wsp + 255) & ~(uintptr_t)255;
    wsp = p + bytes;
    return (void*)p;
  };
  float*    xA    = (float*)alloc(sizeof(float) * ROWS * DIMC);
  float*    xB    = (float*)alloc(sizeof(float) * ROWS * DIMC);
  _Float16* h16   = (_Float16*)alloc(sizeof(_Float16) * ROWS * DIMC);
  _Float16* a16   = (_Float16*)alloc(sizeof(_Float16) * ROWS * 768);
  _Float16* w16   = (_Float16*)alloc(sizeof(_Float16) * (size_t)MLPC * DIMC);
  float*    qkvb  = (float*)alloc(sizeof(float) * ROWS * 3 * DIMC);
  size_t qkmax  = 144ull * WTOK * HDP;                    // >= 16*1024*64
  _Float16* q16   = (_Float16*)alloc(sizeof(_Float16) * qkmax);
  _Float16* k16   = (_Float16*)alloc(sizeof(_Float16) * qkmax);
  size_t vmax   = 144ull * HDP * WTOKP;                   // >= 16*64*1024
  _Float16* v16t  = (_Float16*)alloc(sizeof(_Float16) * vmax);
  size_t atmax  = 16ull * TOK * TOK;                      // >= 144*196*196
  float*    attn  = (float*)alloc(sizeof(float) * atmax);
  size_t at16max = 16ull * TOK * TOK;                     // >= 144*196*224
  _Float16* attn16= (_Float16*)alloc(sizeof(_Float16) * at16max);
  size_t relmax = 16ull * TOK * GRD;                      // >= 144*196*14
  float*    relh  = (float*)alloc(sizeof(float) * relmax);
  float*    relw  = (float*)alloc(sizeof(float) * relmax);
  float*    ctx   = (float*)alloc(sizeof(float) * 144ull * WTOK * HDP);
  float*    f1    = (float*)alloc(sizeof(float) * (size_t)ROWS * MLPC);
  _Float16* f1h   = (_Float16*)alloc(sizeof(_Float16) * (size_t)ROWS * MLPC);
  _Float16* m16   = (_Float16*)alloc(sizeof(_Float16) * ROWS * DIMC);

  auto gemm = [&](const _Float16* A, long long sA, int lda,
                  const _Float16* Bt, long long sB, int ldb,
                  float* C, long long sC, int ldc,
                  _Float16* C16, long long sC16, int ldc16,
                  const float* bias, const float* res,
                  int M, int N, int K, int batch, float alpha, int act) {
    dim3 g((N + 63) / 64, (M + 127) / 128, batch);
    gemm_wmma<<<g, 256, 0, stream>>>(A, sA, lda, Bt, sB, ldb, C, sC, ldc,
                                     C16, sC16, ldc16, bias, res, M, N, K, alpha, act);
  };
  auto grid1 = [](long long n) { return (unsigned)((n + 255) / 256); };

  // -------- patch embed: im2col + GEMM (+bias) + pos --------
  im2col_patch<<<ROWS, 256, 0, stream>>>(X, a16);
  wconv_copy<<<grid1((long long)DIMC * 768), 256, 0, stream>>>(patch_w, w16, (long long)DIMC * 768);
  gemm(a16, 0, 768, w16, 0, 768, xA, 0, DIMC, nullptr, 0, 0,
       patch_b, nullptr, ROWS, DIMC, 768, 1, 1.f, 0);
  add_pos<<<grid1(OUTN), 256, 0, stream>>>(xA, pos);

  float* x  = xA;
  float* xn = xB;
  const float qscale = 0.14433756729740643f;   // 1/sqrt(48)

  for (int i = 0; i < DEPTH; ++i) {
    const bool glob = (i == 2 || i == 8);
    const int T   = glob ? TOK : WTOK;
    const int Tp  = glob ? TOK : WTOKP;        // padded K for attn*V
    const int W   = glob ? GRD : WSZ;
    const int S   = glob ? GRD : WSZ;
    const int NBH = glob ? (BQ * HEADS) : (BQ * NWIN * HEADS);

    // ---- attention ----
    ln_kernel<<<ROWS, 128, 0, stream>>>(x, blkP(i, 0), blkP(i, 1), h16, DIMC);
    wconv_t<<<grid1((long long)3 * DIMC * DIMC), 256, 0, stream>>>(blkP(i, 2), w16, DIMC, 3 * DIMC);
    gemm(h16, 0, DIMC, w16, 0, DIMC, qkvb, 0, 3 * DIMC, nullptr, 0, 0,
         blkP(i, 3), nullptr, ROWS, 3 * DIMC, DIMC, 1, 1.f, 0);

    fill_h16<<<grid1((long long)NBH * HDP * Tp), 256, 0, stream>>>(v16t, (long long)NBH * HDP * Tp);
    if (glob)
      split_qkv_global<<<grid1(16LL * TOK * HDP), 256, 0, stream>>>(qkvb, q16, k16, v16t, Tp);
    else
      split_qkv_win<<<grid1(144LL * WTOK * HDP), 256, 0, stream>>>(qkvb, blkP(i, 3), q16, k16, v16t, Tp);

    // scores = (q*scale) @ k^T   (HDP=64 with zero pad -> exact)
    gemm(q16, (long long)T * HDP, HDP, k16, (long long)T * HDP, HDP,
         attn, (long long)T * T, T, nullptr, 0, 0,
         nullptr, nullptr, T, T, HDP, NBH, qscale, 0);

    rel_bias<<<(unsigned)((long long)NBH * T), 32, 0, stream>>>(q16, blkP(i, 6), relh, T, W, S, 1);
    rel_bias<<<(unsigned)((long long)NBH * T), 32, 0, stream>>>(q16, blkP(i, 7), relw, T, W, S, 0);
    softmax_bias<<<(unsigned)((long long)NBH * T), 256, 0, stream>>>(attn, attn16, relh, relw, T, Tp, W, S);

    // ctx = attn @ v   (K and N fully padded -> B tiles always take the interior path)
    gemm(attn16, (long long)T * Tp, Tp, v16t, (long long)HDP * Tp, Tp,
         ctx, (long long)T * HDP, HDP, nullptr, 0, 0,
         nullptr, nullptr, T, HDP, Tp, NBH, 1.f, 0);

    if (glob) merge_global<<<grid1(OUTN), 256, 0, stream>>>(ctx, m16);
    else      merge_win<<<grid1(OUTN), 256, 0, stream>>>(ctx, m16);

    // proj + residual
    wconv_t<<<grid1((long long)DIMC * DIMC), 256, 0, stream>>>(blkP(i, 4), w16, DIMC, DIMC);
    gemm(m16, 0, DIMC, w16, 0, DIMC, xn, 0, DIMC, nullptr, 0, 0,
         blkP(i, 5), x, ROWS, DIMC, DIMC, 1, 1.f, 0);
    { float* t = x; x = xn; xn = t; }

    // ---- MLP ----
    ln_kernel<<<ROWS, 128, 0, stream>>>(x, blkP(i, 8), blkP(i, 9), h16, DIMC);
    wconv_t<<<grid1((long long)DIMC * MLPC), 256, 0, stream>>>(blkP(i, 10), w16, DIMC, MLPC);
    gemm(h16, 0, DIMC, w16, 0, DIMC, f1, 0, MLPC, f1h, 0, MLPC,
         blkP(i, 11), nullptr, ROWS, MLPC, DIMC, 1, 1.f, 1 /*GELU*/);
    wconv_t<<<grid1((long long)MLPC * DIMC), 256, 0, stream>>>(blkP(i, 12), w16, MLPC, DIMC);
    gemm(f1h, 0, MLPC, w16, 0, MLPC, xn, 0, DIMC, nullptr, 0, 0,
         blkP(i, 13), x, ROWS, DIMC, MLPC, 1, 1.f, 0);
    { float* t = x; x = xn; xn = t; }

    if (i == 0)  copyf<<<grid1(OUTN), 256, 0, stream>>>(x, out + 1 * OUTN, OUTN);
    if (i == 5)  copyf<<<grid1(OUTN), 256, 0, stream>>>(x, out + 2 * OUTN, OUTN);
    if (i == 11) copyf<<<grid1(OUTN), 256, 0, stream>>>(x, out + 3 * OUTN, OUTN);
  }
  copyf<<<grid1(OUTN), 256, 0, stream>>>(x, out, OUTN);
}
